// ChaiMPNN_48146583388296
// MI455X (gfx1250) — matrix operand order repaired and checked by reading, not compiled, not run on hardware
//
#include <hip/hip_runtime.h>
#include <hip/hip_bf16.h>
#include <math.h>

typedef __attribute__((ext_vector_type(16))) __bf16 bf16x16;
typedef __attribute__((ext_vector_type(8)))  float  f32x8;
typedef __attribute__((ext_vector_type(4)))  unsigned int u32x4;
typedef unsigned short u16;
typedef unsigned int   u32;

static constexpr int B_ = 256, K_ = 30, NODE_IN = 384, EDGE_IN = 256, L_ = 3;
static constexpr int N_   = B_ * K_;          // 7680 nodes
static constexpr int EPG  = K_ * (K_ - 1);    // 870 edges per graph
static constexpr int E_   = B_ * EPG;         // 222720 edges
static constexpr float EPS_ = 1e-5f;

// ---------------- helpers ----------------

__device__ inline u16 f2bf(float f) {              // RNE f32 -> bf16
  u32 u = __float_as_uint(f);
  u32 r = u + 0x7FFFu + ((u >> 16) & 1u);
  return (u16)(r >> 16);
}
__device__ inline float gelu_f(float x) {          // exact gelu
  return 0.5f * x * (1.0f + erff(x * 0.70710678118654752f));
}
__device__ inline float wsum(float v) {            // wave32 reduction
  for (int o = 16; o > 0; o >>= 1) v += __shfl_xor(v, o, 32);
  return v;
}

union FragU { bf16x16 v; u32x4 q[2]; };

// A-fragment (16x32 bf16) from row-major u16 buffer. Per-ISA layout:
// lane row M=lane&15; elems 0..7 -> K=kbase..kbase+7, elems 8..15 -> K=kbase+16..+23,
// kbase = 8*(lane>>4). Caller resolves row (supports gathered rows).
__device__ inline bf16x16 load_a(const u16* base, int stride, int row, int k0) {
  const int lane = threadIdx.x & 31;
  const int klo  = (lane >> 4) * 8;
  const u16* p = base + row * stride + k0 + klo;
  FragU f;
  f.q[0] = *(const u32x4*)(p);
  f.q[1] = *(const u32x4*)(p + 16);
  return f.v;
}

// B-fragment (32x16 bf16) from transposed weights wT[N][Ktot] (row-major).
// lane col N=lane&15; elems 0..7 -> K = k0 + 8*(lane>>4) + 0..7, elems 8..15 -> K = k0 + 16 + 8*(lane>>4) + 0..7.
__device__ inline bf16x16 load_b(const u16* wT, int Ktot, int ntile, int k0) {
  const int lane = threadIdx.x & 31;
  const u16* p = wT + (size_t)(ntile * 16 + (lane & 15)) * Ktot + k0 + ((lane >> 4) * 8);
  FragU f;
  f.q[0] = *(const u32x4*)(p);
  f.q[1] = *(const u32x4*)(p + 16);
  return f.v;
}

// A-fragment from an f32 buffer, converting to bf16 on the fly.
__device__ inline bf16x16 load_a_f32(const float* base, int stride, int row, int k0) {
  const int lane = threadIdx.x & 31;
  const int klo  = (lane >> 4) * 8;
  const float* p = base + row * stride + k0 + klo;
  u16 t[16];
#pragma unroll
  for (int i = 0; i < 8; ++i) t[i] = f2bf(p[i]);
#pragma unroll
  for (int i = 0; i < 8; ++i) t[8 + i] = f2bf(p[16 + i]);
  bf16x16 v;
  __builtin_memcpy(&v, t, sizeof(v));
  return v;
}

__device__ inline f32x8 wmma_bf(bf16x16 a, bf16x16 b, f32x8 c) {
  return __builtin_amdgcn_wmma_f32_16x16x32_bf16(false, a, false, b, (short)0, c, false, false);
}

// ---------------- kernel 1: weight convert + transpose (fp32 KxN -> bf16 NxK) ----------------

__global__ void k_prep_wT(const float* __restrict__ w, u16* __restrict__ wT, int Kin, int Nout) {
  int idx = blockIdx.x * blockDim.x + threadIdx.x;
  if (idx >= Kin * Nout) return;
  int n = idx / Kin, k = idx - n * Kin;
  wT[idx] = f2bf(w[(size_t)k * Nout + n]);
}

// ---------------- kernel 2: embed = GELU(LN(x @ W + b)) ----------------
// 16 rows x 128 cols per block (4 waves, 2 n-tiles per wave), K = IN.

template <int IN, bool WF32>
__global__ __launch_bounds__(128) void k_embed(
    const float* __restrict__ x, const u16* __restrict__ wT,
    const float* __restrict__ bias, const float* __restrict__ g, const float* __restrict__ be,
    float* __restrict__ out_f32, u16* __restrict__ out_bf) {
  constexpr int STR = IN + 8;                         // u16 stride, 16B aligned
  __shared__ __align__(16) u16   sx[16 * STR];
  __shared__ __align__(16) float so[16 * 128];
  const int r0  = blockIdx.x * 16;
  const int tid = threadIdx.x;
  for (int i = tid; i < 16 * IN; i += 128) {          // stage + cvt to bf16
    int r = i / IN, c = i - r * IN;
    sx[r * STR + c] = f2bf(x[(size_t)(r0 + r) * IN + c]);
  }
  __syncthreads();
  const int wv = tid >> 5, lane = tid & 31;
  const int mrow = lane & 15;
#pragma unroll
  for (int t = 0; t < 2; ++t) {
    const int nt = wv + t * 4;
    f32x8 acc;
    const float bv = bias[nt * 16 + (lane & 15)];
#pragma unroll
    for (int i = 0; i < 8; ++i) acc[i] = bv;
#pragma unroll
    for (int ks = 0; ks < IN / 32; ++ks) {
      bf16x16 a = load_a(sx, STR, mrow, ks * 32);
      bf16x16 b = load_b(wT, IN, nt, ks * 32);
      acc = wmma_bf(a, b, acc);
    }
#pragma unroll
    for (int i = 0; i < 8; ++i)
      so[(i + 8 * (lane >> 4)) * 128 + nt * 16 + (lane & 15)] = acc[i];
  }
  __syncthreads();
#pragma unroll
  for (int q = 0; q < 4; ++q) {                       // LN + GELU, 1 row per (wave,q)
    const int r = wv * 4 + q;
    float v[4], s = 0.f;
#pragma unroll
    for (int j = 0; j < 4; ++j) { v[j] = so[r * 128 + lane + 32 * j]; s += v[j]; }
    s = wsum(s);
    const float mu = s * (1.0f / 128.0f);
    float s2 = 0.f;
#pragma unroll
    for (int j = 0; j < 4; ++j) { float d = v[j] - mu; s2 += d * d; }
    s2 = wsum(s2);
    const float rstd = rsqrtf(s2 * (1.0f / 128.0f) + EPS_);
#pragma unroll
    for (int j = 0; j < 4; ++j) {
      const int c = lane + 32 * j;
      float y = gelu_f((v[j] - mu) * rstd * g[c] + be[c]);
      size_t o = (size_t)(r0 + r) * 128 + c;
      if constexpr (WF32) out_f32[o] = y;
      out_bf[o] = f2bf(y);
    }
  }
}

// ---------------- kernel 3: fused 3-layer MPNN + readout, one workgroup per graph ----------------

struct FusedArgs {
  const float* h0;                 // N x 128 f32
  const u16*   ea;                 // E x 128 bf16
  const u16*   m1wT[3];            // 128 x 384 bf16^T
  const u16*   m2wT[3];            // 128 x 128
  const u16*   uwT[3];             // 128 x 256
  const float* m1b[3]; const float* m2b[3];
  const float* ub[3];  const float* ug[3]; const float* ube[3];
  const float* h1w; const float* h1b;
  const float* h2w; const float* h2b;
  const float* h3w; const float* h3b;
  float* out;
};

__global__ __launch_bounds__(128) void k_mpnn(FusedArgs A) {
  constexpr int HS = 136;                            // bf16 row stride (16B aligned, conflict pad)
  __shared__ __align__(16) u16   s_hb[K_ * HS];      // h bf16
  __shared__ __align__(16) float s_hf[K_ * 128];     // h f32 (residual)
  __shared__ __align__(16) float s_ag[K_ * 128];     // aggregation
  __shared__ __align__(16) u16   s_ea[16 * HS];      // ea tile
  __shared__ __align__(16) u16   s_m [16 * HS];      // m1 intermediate (bf16)
  __shared__ __align__(16) float s_u [32 * 128];     // u pre-LN
  __shared__ __align__(16) float s_r [256];
  __shared__ __align__(16) float s_t [128];

  const int b   = blockIdx.x;
  const int tid = threadIdx.x;
  const int wv  = tid >> 5, lane = tid & 31;

  for (int i = tid; i < K_ * 128; i += 128) {        // load graph's h
    int r = i >> 7, c = i & 127;
    float v = A.h0[((size_t)b * K_ + r) * 128 + c];
    s_hf[i] = v;
    s_hb[r * HS + c] = f2bf(v);
  }
  __syncthreads();

  for (int l = 0; l < L_; ++l) {
    for (int i = tid; i < K_ * 128; i += 128) s_ag[i] = 0.f;
    __syncthreads();
    const u16*   m1wT = A.m1wT[l];
    const u16*   m2wT = A.m2wT[l];
    const float* m1b  = A.m1b[l];
    const float* m2b  = A.m2b[l];

    for (int t = 0; t < 55; ++t) {                   // 870 edges in tiles of 16
      const int e0 = t * 16;
      for (int i = tid; i < 16 * 128; i += 128) {    // stage ea tile
        int r = i >> 7, c = i & 127;
        int e = e0 + r; e = e < EPG ? e : EPG - 1;
        s_ea[r * HS + c] = A.ea[((size_t)b * EPG + e) * 128 + c];
      }
      if (t + 1 < 55)                                // CDNA5 global_prefetch_b8 of next tile
        __builtin_prefetch(&A.ea[((size_t)b * EPG + e0 + 16) * 128 + (tid & 127)], 0, 1);
      __syncthreads();

      int e = e0 + (lane & 15);                      // per-lane edge decode (branch-free)
      e = e < EPG ? e : EPG - 1;
      const int sl = e / (K_ - 1);
      const int j  = e - sl * (K_ - 1);
      const int dl = j + (j >= sl ? 1 : 0);

#pragma unroll
      for (int half = 0; half < 2; ++half) {         // m1: [xi|xj|ea] @ W1, K=384
        const int nt = wv + half * 4;
        f32x8 acc;
        const float bv = m1b[nt * 16 + (lane & 15)];
#pragma unroll
        for (int i = 0; i < 8; ++i) acc[i] = bv;
#pragma unroll
        for (int ks = 0; ks < 12; ++ks) {
          bf16x16 a;
          if (ks < 4)      a = load_a(s_hb, HS, dl, ks * 32);           // xi = h[dst]
          else if (ks < 8) a = load_a(s_hb, HS, sl, (ks - 4) * 32);     // xj = h[src]
          else             a = load_a(s_ea, HS, lane & 15, (ks - 8) * 32);
          bf16x16 bb = load_b(m1wT, 384, nt, ks * 32);
          acc = wmma_bf(a, bb, acc);
        }
#pragma unroll
        for (int i = 0; i < 8; ++i)
          s_m[(i + 8 * (lane >> 4)) * HS + nt * 16 + (lane & 15)] = f2bf(gelu_f(acc[i]));
      }
      __syncthreads();

#pragma unroll
      for (int half = 0; half < 2; ++half) {         // m2 (K=128) + scatter-add into agg
        const int nt = wv + half * 4;
        f32x8 acc;
        const float bv = m2b[nt * 16 + (lane & 15)];
#pragma unroll
        for (int i = 0; i < 8; ++i) acc[i] = bv;
#pragma unroll
        for (int ks = 0; ks < 4; ++ks) {
          bf16x16 a  = load_a(s_m, HS, lane & 15, ks * 32);
          bf16x16 bb = load_b(m2wT, 128, nt, ks * 32);
          acc = wmma_bf(a, bb, acc);
        }
#pragma unroll
        for (int i = 0; i < 8; ++i) {
          const int r  = i + 8 * (lane >> 4);
          const int ee = e0 + r;
          if (ee < EPG) {
            int ssl = ee / (K_ - 1);
            int jj  = ee - ssl * (K_ - 1);
            int ddl = jj + (jj >= ssl ? 1 : 0);
            atomicAdd(&s_ag[ddl * 128 + nt * 16 + (lane & 15)], acc[i]);  // ds_add_f32
          }
        }
      }
      __syncthreads();
    }

    const u16*   uwT = A.uwT[l];                     // u = [h, agg] @ Wu + bu  (K=256)
    const float* ub  = A.ub[l];
#pragma unroll
    for (int mt = 0; mt < 2; ++mt) {
      const int m0 = mt * 16;
      int row = m0 + (lane & 15);
      int rc  = row < K_ ? row : K_ - 1;
#pragma unroll
      for (int half = 0; half < 2; ++half) {
        const int nt = wv + half * 4;
        f32x8 acc;
        const float bv = ub[nt * 16 + (lane & 15)];
#pragma unroll
        for (int i = 0; i < 8; ++i) acc[i] = bv;
#pragma unroll
        for (int ks = 0; ks < 8; ++ks) {
          bf16x16 a;
          if (ks < 4) a = load_a(s_hb, HS, rc, ks * 32);
          else        a = load_a_f32(s_ag, 128, rc, (ks - 4) * 32);
          bf16x16 bb = load_b(uwT, 256, nt, ks * 32);
          acc = wmma_bf(a, bb, acc);
        }
#pragma unroll
        for (int i = 0; i < 8; ++i)
          s_u[(m0 + i + 8 * (lane >> 4)) * 128 + nt * 16 + (lane & 15)] = acc[i];
      }
    }
    __syncthreads();

    const float* ug = A.ug[l];                       // h = h + LN(u)
    const float* ube = A.ube[l];
    for (int r = wv; r < K_; r += 4) {
      float v[4], s = 0.f;
#pragma unroll
      for (int jx = 0; jx < 4; ++jx) { v[jx] = s_u[r * 128 + lane + 32 * jx]; s += v[jx]; }
      s = wsum(s);
      const float mu = s * (1.f / 128.f);
      float s2 = 0.f;
#pragma unroll
      for (int jx = 0; jx < 4; ++jx) { float d = v[jx] - mu; s2 += d * d; }
      s2 = wsum(s2);
      const float rstd = rsqrtf(s2 * (1.f / 128.f) + EPS_);
#pragma unroll
      for (int jx = 0; jx < 4; ++jx) {
        const int c = lane + 32 * jx;
        float hn = s_hf[r * 128 + c] + (v[jx] - mu) * rstd * ug[c] + ube[c];
        s_hf[r * 128 + c] = hn;
        s_hb[r * HS + c]  = f2bf(hn);
      }
    }
    __syncthreads();
  }

  // readout: r = [h[node0], mean(h)] -> h1(gelu) -> h2(gelu) -> h3
  if (tid < 128) {
    float gsum = 0.f;
    for (int r = 0; r < K_; ++r) gsum += s_hf[r * 128 + tid];
    s_r[tid]       = s_hf[tid];
    s_r[128 + tid] = gsum * (1.0f / K_);
  }
  __syncthreads();
  if (tid < 128) {
    float acc = A.h1b[tid];
    for (int k = 0; k < 256; ++k) acc += s_r[k] * A.h1w[k * 128 + tid];
    s_t[tid] = gelu_f(acc);
  }
  __syncthreads();
  if (tid < 64) {
    float acc = A.h2b[tid];
    for (int k = 0; k < 128; ++k) acc += s_t[k] * A.h2w[k * 64 + tid];
    s_r[tid] = gelu_f(acc);
  }
  __syncthreads();
  if (tid == 0) {
    float acc = A.h3b[0];
    for (int k = 0; k < 64; ++k) acc += s_r[k] * A.h3w[k];
    A.out[b] = acc;
  }
}

// ---------------- host ----------------

extern "C" void kernel_launch(void* const* d_in, const int* in_sizes, int n_in,
                              void* d_out, int out_size, void* d_ws, size_t ws_size,
                              hipStream_t stream) {
  (void)in_sizes; (void)n_in; (void)out_size; (void)ws_size;
  const float* x    = (const float*)d_in[0];
  const float* eatt = (const float*)d_in[1];
  const float* ne_w = (const float*)d_in[2];
  const float* ne_b = (const float*)d_in[3];
  const float* ne_g = (const float*)d_in[4];
  const float* ne_be= (const float*)d_in[5];
  const float* ee_w = (const float*)d_in[6];
  const float* ee_b = (const float*)d_in[7];
  const float* ee_g = (const float*)d_in[8];
  const float* ee_be= (const float*)d_in[9];
  // per-layer params start at 10, 8 tensors each: m1_w,m1_b,m2_w,m2_b,u_w,u_b,u_g,u_be
  const float* h1w = (const float*)d_in[34];
  const float* h1b = (const float*)d_in[35];
  const float* h2w = (const float*)d_in[36];
  const float* h2b = (const float*)d_in[37];
  const float* h3w = (const float*)d_in[38];
  const float* h3b = (const float*)d_in[39];

  char* ws = (char*)d_ws;
  auto carve = [&](size_t bytes) -> char* {
    char* p = ws; ws += (bytes + 255) & ~(size_t)255; return p;
  };
  float* h_f32 = (float*)carve((size_t)N_ * 128 * 4);
  u16*   h_bf  = (u16*)  carve((size_t)N_ * 128 * 2);
  u16*   ea_bf = (u16*)  carve((size_t)E_ * 128 * 2);
  u16*   ne_wT = (u16*)  carve((size_t)128 * 384 * 2);
  u16*   ee_wT = (u16*)  carve((size_t)128 * 256 * 2);
  u16*   m1_wT[3]; u16* m2_wT[3]; u16* u_wT[3];
  for (int l = 0; l < 3; ++l) {
    m1_wT[l] = (u16*)carve((size_t)128 * 384 * 2);
    m2_wT[l] = (u16*)carve((size_t)128 * 128 * 2);
    u_wT[l]  = (u16*)carve((size_t)128 * 256 * 2);
  }

  auto prep = [&](const float* w, u16* wT, int Kin, int Nout) {
    int n = Kin * Nout;
    k_prep_wT<<<(n + 255) / 256, 256, 0, stream>>>(w, wT, Kin, Nout);
  };
  prep(ne_w, ne_wT, 384, 128);
  prep(ee_w, ee_wT, 256, 128);
  for (int l = 0; l < 3; ++l) {
    const float* m1w = (const float*)d_in[10 + 8 * l + 0];
    const float* m2w = (const float*)d_in[10 + 8 * l + 2];
    const float* uw  = (const float*)d_in[10 + 8 * l + 4];
    prep(m1w, m1_wT[l], 384, 128);
    prep(m2w, m2_wT[l], 128, 128);
    prep(uw,  u_wT[l],  256, 128);
  }

  k_embed<NODE_IN, true ><<<N_ / 16, 128, 0, stream>>>(x,    ne_wT, ne_b, ne_g, ne_be, h_f32,  h_bf);
  k_embed<EDGE_IN, false><<<E_ / 16, 128, 0, stream>>>(eatt, ee_wT, ee_b, ee_g, ee_be, nullptr, ea_bf);

  FusedArgs fa{};
  fa.h0 = h_f32;
  fa.ea = ea_bf;
  for (int l = 0; l < 3; ++l) {
    fa.m1wT[l] = m1_wT[l]; fa.m2wT[l] = m2_wT[l]; fa.uwT[l] = u_wT[l];
    fa.m1b[l] = (const float*)d_in[10 + 8 * l + 1];
    fa.m2b[l] = (const float*)d_in[10 + 8 * l + 3];
    fa.ub[l]  = (const float*)d_in[10 + 8 * l + 5];
    fa.ug[l]  = (const float*)d_in[10 + 8 * l + 6];
    fa.ube[l] = (const float*)d_in[10 + 8 * l + 7];
  }
  fa.h1w = h1w; fa.h1b = h1b;
  fa.h2w = h2w; fa.h2b = h2b;
  fa.h3w = h3w; fa.h3b = h3b;
  fa.out = (float*)d_out;

  k_mpnn<<<B_, 128, 0, stream>>>(fa);
}
